// BayesianTransformerBlock_56160992363010
// MI455X (gfx1250) — compile-verified, matrix operand order checked by reading, not run on hardware
//
#include <hip/hip_runtime.h>
#include <hip/hip_bf16.h>
#include <stdint.h>

#define BS   4
#define SLEN 4096
#define DIM  512
#define HID  2048
#define NTOK (BS*SLEN)

typedef __bf16 bf16;
typedef __attribute__((ext_vector_type(8)))  bf16  v8bf;
typedef __attribute__((ext_vector_type(16))) bf16  v16bf;
typedef __attribute__((ext_vector_type(8)))  float v8f;

union Frag { v16bf v; v8bf h[2]; };

__device__ __forceinline__ bf16 f2bf(float f) {
  union { float f; uint32_t u; } v; v.f = f;
  uint32_t r = v.u + 0x7fffu + ((v.u >> 16) & 1u);   // round-to-nearest-even
  uint16_t h = (uint16_t)(r >> 16);
  return __builtin_bit_cast(bf16, h);
}

__device__ __forceinline__ v8f wmma_bf16(const Frag& a, const Frag& b, v8f c) {
  // D = A(16x32) * B(32x16) + C, f32 accumulate
  return __builtin_amdgcn_wmma_f32_16x16x32_bf16(false, a.v, false, b.v,
                                                 (short)0, c, false, false);
}

// ---------------------------------------------------------------------------
// W = mu + exp(ls) * eps  ->  bf16
// ---------------------------------------------------------------------------
__global__ void bayes_weight_kernel(const float* __restrict__ mu,
                                    const float* __restrict__ ls,
                                    const float* __restrict__ eps,
                                    bf16* __restrict__ out, int n) {
  int i = blockIdx.x * blockDim.x + threadIdx.x;
  if (i < n) out[i] = f2bf(mu[i] + __expf(ls[i]) * eps[i]);
}

__global__ void cvt_bf16_kernel(const float* __restrict__ in,
                                bf16* __restrict__ out, int n) {
  int i = blockIdx.x * blockDim.x + threadIdx.x;
  if (i < n) out[i] = f2bf(in[i]);
}

// ---------------------------------------------------------------------------
// Generic WMMA GEMM: Y[n,o] = sum_d X[n,d] * W[o,d]   (both row-major bf16)
// Block: 256 threads / 8 waves, 128x128 tile; wave -> 32(M) x 64(N) subtile.
// mode 0: Y -> bf16 row-major
// mode 1: Y -> bf16 transposed per batch:  outB[(b*O + o)*SLEN + t]
// mode 2: Y + resF -> outF (f32) and outB (bf16)          (wo + residual)
// mode 3: relu(Y) -> bf16 row-major                        (w1)
// mode 4: Y + resF -> outF (f32)                           (w2 + residual)
// ---------------------------------------------------------------------------
__global__ __launch_bounds__(256)
void gemm_wmma_kernel(const bf16* __restrict__ X, const bf16* __restrict__ Wt,
                      int D, int O,
                      bf16* __restrict__ outB, float* __restrict__ outF,
                      const float* __restrict__ resF, int mode) {
  const int tid  = threadIdx.x;
  const int w    = tid >> 5;
  const int lane = tid & 31;
  const int lh   = lane & 15;      // lane within 16-half
  const int hi   = lane >> 4;      // which half (K/M offset selector)

  const int m0 = blockIdx.x * 128 + (w & 3) * 32;   // 2 M sub-tiles of 16
  const int n0 = blockIdx.y * 128 + (w >> 2) * 64;  // 4 N sub-tiles of 16

  v8f acc[2][4];
#pragma unroll
  for (int mi = 0; mi < 2; ++mi)
#pragma unroll
    for (int ni = 0; ni < 4; ++ni) acc[mi][ni] = (v8f){};

  const bf16* xr0 = X + (size_t)(m0 + lh) * D;
  const bf16* xr1 = X + (size_t)(m0 + 16 + lh) * D;
  const bf16* wr0 = Wt + (size_t)(n0 +  0 + lh) * D;
  const bf16* wr1 = Wt + (size_t)(n0 + 16 + lh) * D;
  const bf16* wr2 = Wt + (size_t)(n0 + 32 + lh) * D;
  const bf16* wr3 = Wt + (size_t)(n0 + 48 + lh) * D;

  for (int k0 = 0; k0 < D; k0 += 32) {
    const int klo = k0 + 8 * hi;
    const int khi = k0 + 16 + 8 * hi;
    Frag a[2], b[4];
    a[0].h[0] = *(const v8bf*)&xr0[klo];  a[0].h[1] = *(const v8bf*)&xr0[khi];
    a[1].h[0] = *(const v8bf*)&xr1[klo];  a[1].h[1] = *(const v8bf*)&xr1[khi];
    b[0].h[0] = *(const v8bf*)&wr0[klo];  b[0].h[1] = *(const v8bf*)&wr0[khi];
    b[1].h[0] = *(const v8bf*)&wr1[klo];  b[1].h[1] = *(const v8bf*)&wr1[khi];
    b[2].h[0] = *(const v8bf*)&wr2[klo];  b[2].h[1] = *(const v8bf*)&wr2[khi];
    b[3].h[0] = *(const v8bf*)&wr3[klo];  b[3].h[1] = *(const v8bf*)&wr3[khi];
#pragma unroll
    for (int mi = 0; mi < 2; ++mi)
#pragma unroll
      for (int ni = 0; ni < 4; ++ni)
        acc[mi][ni] = wmma_bf16(a[mi], b[ni], acc[mi][ni]);
  }

  // Epilogue. C layout: VGPR r, lanes[0:15] -> M=r, lanes[16:31] -> M=r+8.
  if (mode == 1) {
    // transposed store, per lane 8 consecutive tokens -> one b128 store
#pragma unroll
    for (int mi = 0; mi < 2; ++mi)
#pragma unroll
      for (int ni = 0; ni < 4; ++ni) {
        const int rowb = m0 + mi * 16 + 8 * hi;   // 8 consecutive rows
        const int col  = n0 + ni * 16 + lh;
        const int batch = rowb >> 12;             // 4096 tokens per batch
        const int tloc  = rowb & (SLEN - 1);
        v8bf pk;
#pragma unroll
        for (int r = 0; r < 8; ++r) pk[r] = f2bf(acc[mi][ni][r]);
        *(v8bf*)&outB[((size_t)batch * O + col) * SLEN + tloc] = pk;
      }
  } else {
#pragma unroll
    for (int mi = 0; mi < 2; ++mi)
#pragma unroll
      for (int ni = 0; ni < 4; ++ni)
#pragma unroll
        for (int r = 0; r < 8; ++r) {
          const int row = m0 + mi * 16 + 8 * hi + r;
          const int col = n0 + ni * 16 + lh;
          const size_t idx = (size_t)row * O + col;
          float v = acc[mi][ni][r];
          if (mode == 0) {
            outB[idx] = f2bf(v);
          } else if (mode == 2) {
            float rv = resF[idx] + v;
            outF[idx] = rv;
            outB[idx] = f2bf(rv);
          } else if (mode == 3) {
            outB[idx] = f2bf(v > 0.f ? v : 0.f);
          } else { // mode 4
            outF[idx] = resF[idx] + v;
          }
        }
  }
}

// ---------------------------------------------------------------------------
// Flash attention, causal, single head, D=512.
// Block: 128 threads / 4 waves, 16 query rows per block.
// scores phase: wave w owns keys [j+16w, j+16w+16)  (1 subtile, 16 WMMAs)
// P.V   phase: wave w owns output D slice [128w, 128w+128)
// ---------------------------------------------------------------------------
__global__ __launch_bounds__(128)
void attn_kernel(const bf16* __restrict__ xq, const bf16* __restrict__ xk,
                 const bf16* __restrict__ xvT, bf16* __restrict__ h) {
  const int b  = blockIdx.y;
  const int q0 = blockIdx.x * 16;
  const int tid  = threadIdx.x;
  const int w    = tid >> 5;
  const int lane = tid & 31;
  const int lh   = lane & 15;
  const int hi   = lane >> 4;

  __shared__ __align__(16) bf16 qtile[16 * DIM];   // 16 KB
  __shared__ __align__(16) bf16 ptile[16 * 64];    // 2 KB
  __shared__ float redmax[4][16];
  __shared__ float redsum[4][16];

  const bf16* xqb = xq  + (size_t)b * SLEN * DIM;
  const bf16* xkb = xk  + (size_t)b * SLEN * DIM;
  const bf16* xvb = xvT + (size_t)b * DIM * SLEN;

  // stage Q tile (reused across the whole key loop)
  for (int i = tid; i < 16 * DIM / 8; i += 128)
    *(v8bf*)&qtile[i * 8] = *(const v8bf*)&xqb[(size_t)q0 * DIM + i * 8];
  __syncthreads();

  float runm[8], runl[8];
#pragma unroll
  for (int r = 0; r < 8; ++r) { runm[r] = -3.0e38f; runl[r] = 0.f; }
  v8f O[8];
#pragma unroll
  for (int t = 0; t < 8; ++t) O[t] = (v8f){};

  const float sc = 0.044194173824159216f;  // 1/sqrt(512)
  const int jend = q0 + 16;                // causal: keys < q0+16 suffice

  for (int j = 0; j < jend; j += 64) {
    // ---- scores: S = Q . K^T for keys [j+16w, j+16w+16) ----
    v8f s = (v8f){};
    const int key = j + 16 * w + lh;             // this lane's key (B column)
    const bf16* krow = xkb + (size_t)key * DIM;
    for (int k0 = 0; k0 < DIM; k0 += 32) {
      Frag a, bb;
      a.h[0]  = *(const v8bf*)&qtile[lh * DIM + k0 + 8 * hi];
      a.h[1]  = *(const v8bf*)&qtile[lh * DIM + k0 + 16 + 8 * hi];
      bb.h[0] = *(const v8bf*)&krow[k0 + 8 * hi];
      bb.h[1] = *(const v8bf*)&krow[k0 + 16 + 8 * hi];
      s = wmma_bf16(a, bb, s);
    }
    // scale + causal mask (bitwise identical to reference's -1e9 additive mask)
#pragma unroll
    for (int r = 0; r < 8; ++r) {
      const int m = q0 + r + 8 * hi;
      s[r] = s[r] * sc + (key > m ? -1.0e9f : 0.f);
    }

    // ---- local row max (16-lane half xor reduce) ----
    float pm[8];
#pragma unroll
    for (int r = 0; r < 8; ++r) pm[r] = s[r];
    for (int off = 1; off < 16; off <<= 1)
#pragma unroll
      for (int r = 0; r < 8; ++r) pm[r] = fmaxf(pm[r], __shfl_xor(pm[r], off, 32));
    if (lh == 0)
#pragma unroll
      for (int r = 0; r < 8; ++r) redmax[w][r + 8 * hi] = pm[r];
    __syncthreads();

    // ---- combined new max, online rescale factors ----
    float esc[8], nm[8];
#pragma unroll
    for (int r = 0; r < 8; ++r) {
      const int row = r + 8 * hi;
      float cm = fmaxf(fmaxf(redmax[0][row], redmax[1][row]),
                       fmaxf(redmax[2][row], redmax[3][row]));
      nm[r]  = fmaxf(runm[r], cm);
      esc[r] = __expf(runm[r] - nm[r]);
      runm[r] = nm[r];
    }

    // ---- p = exp(s - newmax), local row sum, publish P to LDS ----
    float ps[8];
#pragma unroll
    for (int r = 0; r < 8; ++r) { s[r] = __expf(s[r] - nm[r]); ps[r] = s[r]; }
    for (int off = 1; off < 16; off <<= 1)
#pragma unroll
      for (int r = 0; r < 8; ++r) ps[r] += __shfl_xor(ps[r], off, 32);
    if (lh == 0)
#pragma unroll
      for (int r = 0; r < 8; ++r) redsum[w][r + 8 * hi] = ps[r];
#pragma unroll
    for (int r = 0; r < 8; ++r)
      ptile[(r + 8 * hi) * 64 + 16 * w + lh] = f2bf(s[r]);
    __syncthreads();

#pragma unroll
    for (int r = 0; r < 8; ++r) {
      const int row = r + 8 * hi;
      runl[r] = runl[r] * esc[r] +
                (redsum[0][row] + redsum[1][row] + redsum[2][row] + redsum[3][row]);
    }
    // rescale running O
#pragma unroll
    for (int t = 0; t < 8; ++t)
#pragma unroll
      for (int r = 0; r < 8; ++r) O[t][r] *= esc[r];

    // ---- P.V over this wave's D slice; A from LDS, B from xv^T (contiguous) ----
    Frag pa0, pa1;
    pa0.h[0] = *(const v8bf*)&ptile[lh * 64 + 0  + 8 * hi];
    pa0.h[1] = *(const v8bf*)&ptile[lh * 64 + 16 + 8 * hi];
    pa1.h[0] = *(const v8bf*)&ptile[lh * 64 + 32 + 8 * hi];
    pa1.h[1] = *(const v8bf*)&ptile[lh * 64 + 48 + 8 * hi];
#pragma unroll
    for (int t = 0; t < 8; ++t) {
      const int d = w * 128 + t * 16 + lh;
      const bf16* vrow = xvb + (size_t)d * SLEN + j;
      Frag b0, b1;
      b0.h[0] = *(const v8bf*)&vrow[0  + 8 * hi];
      b0.h[1] = *(const v8bf*)&vrow[16 + 8 * hi];
      b1.h[0] = *(const v8bf*)&vrow[32 + 8 * hi];
      b1.h[1] = *(const v8bf*)&vrow[48 + 8 * hi];
      O[t] = wmma_bf16(pa0, b0, O[t]);
      O[t] = wmma_bf16(pa1, b1, O[t]);
    }
  }

  // ---- normalize and store h (bf16 row-major) ----
  float rinv[8];
#pragma unroll
  for (int r = 0; r < 8; ++r) rinv[r] = 1.f / runl[r];
#pragma unroll
  for (int t = 0; t < 8; ++t)
#pragma unroll
    for (int r = 0; r < 8; ++r) {
      const int tok = q0 + r + 8 * hi;
      const int d   = w * 128 + t * 16 + lh;
      h[((size_t)b * SLEN + tok) * DIM + d] = f2bf(O[t][r] * rinv[r]);
    }
}

// ---------------------------------------------------------------------------
extern "C" void kernel_launch(void* const* d_in, const int* in_sizes, int n_in,
                              void* d_out, int out_size, void* d_ws, size_t ws_size,
                              hipStream_t stream) {
  (void)in_sizes; (void)n_in; (void)out_size; (void)ws_size;
  const float* x     = (const float*)d_in[0];
  // d_in[1] = mask (synthesized in-kernel: exact -1e9 causal add)
  const float* wk_mu = (const float*)d_in[2],  *wk_ls = (const float*)d_in[3],  *wk_eps = (const float*)d_in[4];
  const float* wv_mu = (const float*)d_in[5],  *wv_ls = (const float*)d_in[6],  *wv_eps = (const float*)d_in[7];
  const float* wo_mu = (const float*)d_in[8],  *wo_ls = (const float*)d_in[9],  *wo_eps = (const float*)d_in[10];
  const float* w1_mu = (const float*)d_in[11], *w1_ls = (const float*)d_in[12], *w1_eps = (const float*)d_in[13];
  const float* w2_mu = (const float*)d_in[14], *w2_ls = (const float*)d_in[15], *w2_eps = (const float*)d_in[16];
  float* out = (float*)d_out;

  char* ws = (char*)d_ws;
  size_t off = 0;
  auto alloc = [&](size_t bytes) -> void* {
    void* p = ws + off;
    off = (off + bytes + 255) & ~(size_t)255;
    return p;
  };
  bf16*  wkb   = (bf16*)alloc((size_t)DIM * DIM * 2);
  bf16*  wvb   = (bf16*)alloc((size_t)DIM * DIM * 2);
  bf16*  wob   = (bf16*)alloc((size_t)DIM * DIM * 2);
  bf16*  w1b   = (bf16*)alloc((size_t)HID * DIM * 2);
  bf16*  w2b   = (bf16*)alloc((size_t)DIM * HID * 2);
  bf16*  xb    = (bf16*)alloc((size_t)NTOK * DIM * 2);
  bf16*  xkb   = (bf16*)alloc((size_t)NTOK * DIM * 2);
  bf16*  xvT   = (bf16*)alloc((size_t)NTOK * DIM * 2);   // [B][DIM][SLEN]
  bf16*  hb    = (bf16*)alloc((size_t)NTOK * DIM * 2);
  float* hresf = (float*)alloc((size_t)NTOK * DIM * 4);
  bf16*  hresb = (bf16*)alloc((size_t)NTOK * DIM * 2);
  bf16*  ff1b  = (bf16*)alloc((size_t)NTOK * HID * 2);

  // 1) materialize Bayesian weights in bf16
  const int TB = 256;
  int nw = DIM * DIM;
  bayes_weight_kernel<<<(nw + TB - 1) / TB, TB, 0, stream>>>(wk_mu, wk_ls, wk_eps, wkb, nw);
  bayes_weight_kernel<<<(nw + TB - 1) / TB, TB, 0, stream>>>(wv_mu, wv_ls, wv_eps, wvb, nw);
  bayes_weight_kernel<<<(nw + TB - 1) / TB, TB, 0, stream>>>(wo_mu, wo_ls, wo_eps, wob, nw);
  nw = HID * DIM;
  bayes_weight_kernel<<<(nw + TB - 1) / TB, TB, 0, stream>>>(w1_mu, w1_ls, w1_eps, w1b, nw);
  bayes_weight_kernel<<<(nw + TB - 1) / TB, TB, 0, stream>>>(w2_mu, w2_ls, w2_eps, w2b, nw);
  int nx = NTOK * DIM;
  cvt_bf16_kernel<<<(nx + TB - 1) / TB, TB, 0, stream>>>(x, xb, nx);

  // 2) xk = x.Wk^T (bf16), xv = x.Wv^T stored transposed per batch
  dim3 g512(NTOK / 128, DIM / 128);
  gemm_wmma_kernel<<<g512, 256, 0, stream>>>(xb, wkb, DIM, DIM, xkb, nullptr, nullptr, 0);
  gemm_wmma_kernel<<<g512, 256, 0, stream>>>(xb, wvb, DIM, DIM, xvT, nullptr, nullptr, 1);

  // 3) causal flash attention -> h (bf16)
  attn_kernel<<<dim3(SLEN / 16, BS), 128, 0, stream>>>(xb, xkb, xvT, hb);

  // 4) h_res = x + h.Wo^T  -> f32 + bf16
  gemm_wmma_kernel<<<g512, 256, 0, stream>>>(hb, wob, DIM, DIM, hresb, hresf, x, 2);

  // 5) ff1 = relu(h_res.W1^T) -> bf16
  dim3 gHid(NTOK / 128, HID / 128);
  gemm_wmma_kernel<<<gHid, 256, 0, stream>>>(hresb, w1b, DIM, HID, ff1b, nullptr, nullptr, 3);

  // 6) out = h_res + ff1.W2^T -> f32
  gemm_wmma_kernel<<<g512, 256, 0, stream>>>(ff1b, w2b, HID, DIM, nullptr, out, hresf, 4);
}
